// Backward_12094627905824
// MI455X (gfx1250) — matrix operand order repaired and checked by reading, not compiled, not run on hardware
//
#include <hip/hip_runtime.h>
#include <hip/hip_bf16.h>
#include <math.h>

typedef _Float16 half_t;
typedef __attribute__((ext_vector_type(16))) _Float16 v16h;
typedef __attribute__((ext_vector_type(8)))  _Float16 v8h;
typedef __attribute__((ext_vector_type(8)))  float    v8f;

union V16 { v16h v; v8h h[2]; };

#define B_TOTAL 262144
#define GCOMP 25
#define H1 128
#define H2 256
#define H3 200
#define H3P 224      // head K padded to multiple of 32
#define NHEAD 300
#define NHEADP 304   // head N padded to multiple of 16

// packed-fragment grid per layer (fragment = 16N x 32K f16 = 512 halves = 1KB)
#define L1_NC 8
#define L1_KC 1
#define L2_NC 16
#define L2_KC 4
#define L3_NC 13
#define L3_KC 8
#define HD_NC 19
#define HD_KC 7

#define L1_OFF 0
#define L1_SZ  (L1_NC*L1_KC*512)
#define L2_OFF (L1_OFF + L1_SZ)
#define L2_SZ  (L2_NC*L2_KC*512)
#define L3_OFF (L2_OFF + L2_SZ)
#define L3_SZ  (L3_NC*L3_KC*512)
#define HD_OFF (L3_OFF + L3_SZ)
#define HD_SZ  (HD_NC*HD_KC*512)

// ---------------------------------------------------------------------------
// Weight packing: B-fragment layout used here, contiguous k order (B 32x16 f16):
//   lane L: n = L%16, kg = L/16; element j (0..15) = B(k = kc*32 + kg*16 + j, n)
//   stored contiguously: dst[frag*512 + lane*16 + j]
// ---------------------------------------------------------------------------
__global__ void pack_rowmajor(const float* __restrict__ src, half_t* __restrict__ dst,
                              int N, int K, int NC, int KC)
{
    int gid = blockIdx.x * blockDim.x + threadIdx.x;
    int total = NC * KC * 32;
    if (gid >= total) return;
    int lane = gid & 31;
    int f    = gid >> 5;
    int nc = f / KC, kc = f % KC;
    int n  = nc * 16 + (lane & 15);
    int kg = lane >> 4;
    half_t* o = dst + (size_t)f * 512 + lane * 16;
    #pragma unroll
    for (int j = 0; j < 16; ++j) {
        int k = kc * 32 + kg * 16 + j;
        float v = (n < N && k < K) ? src[(size_t)n * K + k] : 0.0f;
        o[j] = (half_t)v;
    }
}

__global__ void pack_heads(const float* __restrict__ Wmu, const float* __restrict__ Wsig,
                           const float* __restrict__ Wpai, half_t* __restrict__ dst)
{
    int gid = blockIdx.x * blockDim.x + threadIdx.x;
    int total = HD_NC * HD_KC * 32;
    if (gid >= total) return;
    int lane = gid & 31;
    int f    = gid >> 5;
    int nc = f / HD_KC, kc = f % HD_KC;
    int n  = nc * 16 + (lane & 15);
    int kg = lane >> 4;
    half_t* o = dst + (size_t)f * 512 + lane * 16;
    #pragma unroll
    for (int j = 0; j < 16; ++j) {
        int k = kc * 32 + kg * 16 + j;
        float v = 0.0f;
        if (n < NHEAD && k < H3) {
            int head = n / 100, r = n % 100;                // r = g*4 + d
            const float* s = (head == 0) ? Wmu : (head == 1) ? Wsig : Wpai;
            v = s[(size_t)r * H3 + k];
        }
        o[j] = (half_t)v;
    }
}

// ---------------------------------------------------------------------------
// Fused MDN kernel: 128 rows/block, 8 waves, 1 row-strip of 16 per wave.
// ---------------------------------------------------------------------------
__global__ __launch_bounds__(256)
void mdn_fused(const float* __restrict__ x0, const float* __restrict__ rnd,
               const float* __restrict__ gumbel,
               const half_t* __restrict__ Wp,
               const float* __restrict__ b1, const float* __restrict__ b2,
               const float* __restrict__ b3,
               const float* __restrict__ bmu, const float* __restrict__ bsig,
               const float* __restrict__ bpai,
               float* __restrict__ out)
{
    extern __shared__ __align__(16) char smem[];
    half_t* hA = (half_t*)smem;                 // h1 [128][128] then h3 [128][224]
    half_t* hB = (half_t*)(smem + 65536);       // h2 [128][256]
    half_t* ho = hB;                            // head outputs [128][304] (reuse)

    const int tid  = threadIdx.x;
    const int wave = tid >> 5;
    const int lane = tid & 31;
    const int m    = lane & 15;
    const int kh   = lane >> 4;
    const int row0 = blockIdx.x * 128;
    const int srow = wave * 16;

    // ---------------- layer 1: relu(x0 @ W1^T + b1), K=3 padded to 32
    V16 a1;
    #pragma unroll
    for (int j = 0; j < 16; ++j) a1.v[j] = (half_t)0.0f;
    if (kh == 0) {
        const float* xr = x0 + (size_t)(row0 + srow + m) * 3;
        a1.v[0] = (half_t)xr[0];
        a1.v[1] = (half_t)xr[1];
        a1.v[2] = (half_t)xr[2];
    }
    #pragma unroll
    for (int nc = 0; nc < L1_NC; ++nc) {
        const half_t* bp = Wp + L1_OFF + (size_t)nc * 512 + lane * 16;
        V16 bf; bf.h[0] = *(const v8h*)bp; bf.h[1] = *(const v8h*)(bp + 8);
        float bias = b1[nc * 16 + m];
        v8f c = {bias, bias, bias, bias, bias, bias, bias, bias};
        c = __builtin_amdgcn_wmma_f32_16x16x32_f16(false, a1.v, false, bf.v,
                                                   (short)0, c, false, false);
        #pragma unroll
        for (int v = 0; v < 8; ++v) {
            float r = c[v] > 0.0f ? c[v] : 0.0f;
            hA[(srow + v + 8 * kh) * H1 + nc * 16 + m] = (half_t)r;
        }
    }
    __syncthreads();

    // ---------------- layer 2: relu(h1 @ W2^T + b2)  (K=128, N=256)
    V16 a2[L2_KC];
    #pragma unroll
    for (int kc = 0; kc < L2_KC; ++kc) {
        const half_t* ap = hA + (srow + m) * H1 + kc * 32 + kh * 16;
        a2[kc].h[0] = *(const v8h*)ap;
        a2[kc].h[1] = *(const v8h*)(ap + 8);
    }
    for (int nc = 0; nc < L2_NC; ++nc) {
        float bias = b2[nc * 16 + m];
        v8f c = {bias, bias, bias, bias, bias, bias, bias, bias};
        #pragma unroll
        for (int kc = 0; kc < L2_KC; ++kc) {
            const half_t* bp = Wp + L2_OFF + (size_t)(nc * L2_KC + kc) * 512 + lane * 16;
            V16 bf; bf.h[0] = *(const v8h*)bp; bf.h[1] = *(const v8h*)(bp + 8);
            c = __builtin_amdgcn_wmma_f32_16x16x32_f16(false, a2[kc].v, false, bf.v,
                                                       (short)0, c, false, false);
        }
        #pragma unroll
        for (int v = 0; v < 8; ++v) {
            float r = c[v] > 0.0f ? c[v] : 0.0f;
            hB[(srow + v + 8 * kh) * H2 + nc * 16 + m] = (half_t)r;
        }
    }
    __syncthreads();

    // ---------------- layer 3: relu(h2 @ W3^T + b3)  (K=256, N=200 pad 208)
    V16 a3[L3_KC];
    #pragma unroll
    for (int kc = 0; kc < L3_KC; ++kc) {
        const half_t* ap = hB + (srow + m) * H2 + kc * 32 + kh * 16;
        a3[kc].h[0] = *(const v8h*)ap;
        a3[kc].h[1] = *(const v8h*)(ap + 8);
    }
    for (int nc = 0; nc < L3_NC; ++nc) {
        int n = nc * 16 + m;
        float bias = (n < H3) ? b3[n] : 0.0f;
        v8f c = {bias, bias, bias, bias, bias, bias, bias, bias};
        #pragma unroll
        for (int kc = 0; kc < L3_KC; ++kc) {
            const half_t* bp = Wp + L3_OFF + (size_t)(nc * L3_KC + kc) * 512 + lane * 16;
            V16 bf; bf.h[0] = *(const v8h*)bp; bf.h[1] = *(const v8h*)(bp + 8);
            c = __builtin_amdgcn_wmma_f32_16x16x32_f16(false, a3[kc].v, false, bf.v,
                                                       (short)0, c, false, false);
        }
        #pragma unroll
        for (int v = 0; v < 8; ++v) {
            float r = c[v] > 0.0f ? c[v] : 0.0f;
            hA[(srow + v + 8 * kh) * H3P + n] = (half_t)r;
        }
    }
    // zero-pad h3 columns 208..223 (k-pad for heads)
    {
        int r = tid >> 1;
        int off = 208 + (tid & 1) * 8;
        v8h z = {};
        *(v8h*)&hA[r * H3P + off] = z;
    }
    __syncthreads();

    // ---------------- heads: h3 @ [Wmu;Wsig;Wpai]^T + bias (K=224, N=304)
    V16 ah[HD_KC];
    #pragma unroll
    for (int kc = 0; kc < HD_KC; ++kc) {
        const half_t* ap = hA + (srow + m) * H3P + kc * 32 + kh * 16;
        ah[kc].h[0] = *(const v8h*)ap;
        ah[kc].h[1] = *(const v8h*)(ap + 8);
    }
    for (int nc = 0; nc < HD_NC; ++nc) {
        int n = nc * 16 + m;
        float bias = 0.0f;
        if      (n < 100) bias = bmu[n];
        else if (n < 200) bias = bsig[n - 100];
        else if (n < 300) bias = bpai[n - 200];
        v8f c = {bias, bias, bias, bias, bias, bias, bias, bias};
        #pragma unroll
        for (int kc = 0; kc < HD_KC; ++kc) {
            const half_t* bp = Wp + HD_OFF + (size_t)(nc * HD_KC + kc) * 512 + lane * 16;
            V16 bf; bf.h[0] = *(const v8h*)bp; bf.h[1] = *(const v8h*)(bp + 8);
            c = __builtin_amdgcn_wmma_f32_16x16x32_f16(false, ah[kc].v, false, bf.v,
                                                       (short)0, c, false, false);
        }
        #pragma unroll
        for (int v = 0; v < 8; ++v)
            ho[(srow + v + 8 * kh) * NHEADP + n] = (half_t)c[v];
    }
    __syncthreads();

    // ---------------- Gumbel-argmax selection + affine sample (2 threads/row)
    {
        const int row = tid >> 1;
        const int p   = tid & 1;
        const half_t* hr = ho + row * NHEADP;
        const size_t gb = (size_t)(row0 + row);
        #pragma unroll
        for (int dd = 0; dd < 2; ++dd) {
            const int d = p * 2 + dd;
            float best = -1e30f; int bi = 0;
            for (int g = 0; g < GCOMP; ++g) {
                float pv = fabsf((float)hr[200 + g * 4 + d]);
                float s  = logf(pv + 1e-12f) + gumbel[gb * 100 + g * 4 + d];
                if (s > best) { best = s; bi = g; }
            }
            float mu = (float)hr[bi * 4 + d];
            float sg = fabsf((float)hr[100 + bi * 4 + d]);
            out[gb * 4 + d] = rnd[gb * 4 + d] * sg + mu;
        }
    }
}

// ---------------------------------------------------------------------------
extern "C" void kernel_launch(void* const* d_in, const int* in_sizes, int n_in,
                              void* d_out, int out_size, void* d_ws, size_t ws_size,
                              hipStream_t stream)
{
    const float* x0   = (const float*)d_in[0];
    const float* rnd  = (const float*)d_in[1];
    const float* gum  = (const float*)d_in[2];
    const float* W1   = (const float*)d_in[3];
    const float* b1   = (const float*)d_in[4];
    const float* W2   = (const float*)d_in[5];
    const float* b2   = (const float*)d_in[6];
    const float* W3   = (const float*)d_in[7];
    const float* b3   = (const float*)d_in[8];
    const float* Wmu  = (const float*)d_in[9];
    const float* bmu  = (const float*)d_in[10];
    const float* Wsig = (const float*)d_in[11];
    const float* bsig = (const float*)d_in[12];
    const float* Wpai = (const float*)d_in[13];
    const float* bpai = (const float*)d_in[14];
    float*  out = (float*)d_out;
    half_t* Wp  = (half_t*)d_ws;   // ~316 KB of packed f16 fragments

    pack_rowmajor<<<(L1_NC * L1_KC * 32 + 255) / 256, 256, 0, stream>>>(W1, Wp + L1_OFF, H1, 3,   L1_NC, L1_KC);
    pack_rowmajor<<<(L2_NC * L2_KC * 32 + 255) / 256, 256, 0, stream>>>(W2, Wp + L2_OFF, H2, H1,  L2_NC, L2_KC);
    pack_rowmajor<<<(L3_NC * L3_KC * 32 + 255) / 256, 256, 0, stream>>>(W3, Wp + L3_OFF, H3, H2,  L3_NC, L3_KC);
    pack_heads   <<<(HD_NC * HD_KC * 32 + 255) / 256, 256, 0, stream>>>(Wmu, Wsig, Wpai, Wp + HD_OFF);

    size_t smem = 65536 + (size_t)128 * NHEADP * sizeof(half_t);   // 143360 B
    mdn_fused<<<dim3(B_TOTAL / 128), 256, smem, stream>>>(
        x0, rnd, gum, Wp, b1, b2, b3, bmu, bsig, bpai, out);
}
